// Attention_23364622090354
// MI455X (gfx1250) — matrix operand order repaired and checked by reading, not compiled, not run on hardware
//
#include <hip/hip_runtime.h>

// ---------------------------------------------------------------------------
// Fused GQA attention for MI455X (gfx1250, wave32, WMMA + async/TDM staging,
// double-buffered tile pipelines). B=2, S=2048, DIM=2048, H=16, KV=4, HD=128.
// ---------------------------------------------------------------------------

typedef __attribute__((ext_vector_type(16))) _Float16 v16h;
typedef __attribute__((ext_vector_type(8)))  float    v8f;
typedef unsigned int u32x4 __attribute__((ext_vector_type(4)));
typedef int          i32x4 __attribute__((ext_vector_type(4)));
typedef int          i32x8 __attribute__((ext_vector_type(8)));

#define WMMA_F32_F16(A, Bf, C) \
  __builtin_amdgcn_wmma_f32_16x16x32_f16(false, (A), false, (Bf), (short)0, (C), false, false)

// ---- Async global->LDS (ASYNCcnt path), guarded --------------------------
#if __has_builtin(__builtin_amdgcn_global_load_async_to_lds_b128) && \
    __has_builtin(__builtin_amdgcn_s_wait_asynccnt)
#define HAS_ASYNC_LDS 1
typedef __attribute__((address_space(1))) i32x4 gvec4_t;   // global v4i32
typedef __attribute__((address_space(3))) i32x4 lvec4_t;   // LDS v4i32
__device__ __forceinline__ void async_copy16(const void* g, void* lds) {
  __builtin_amdgcn_global_load_async_to_lds_b128(
      (gvec4_t*)g, (lvec4_t*)lds, 0, 0);
}
#else
#define HAS_ASYNC_LDS 0
#endif

// ---- Tensor Data Mover (TENSORcnt path), guarded -------------------------
#if __has_builtin(__builtin_amdgcn_tensor_load_to_lds) && \
    __has_builtin(__builtin_amdgcn_s_wait_tensorcnt)
#define HAS_TDM 1
typedef __attribute__((address_space(3))) _Float16 lds_f16_t;
// 2D tile load: (tile1 rows x tile0 elems) of 2-byte data from a tensor with
// row stride stride0 (elements), into contiguous LDS at lds_addr.
__device__ __forceinline__ void tdm_load_2d(unsigned lds_addr, const void* gaddr,
                                            unsigned tdim0, unsigned tdim1,
                                            unsigned tile0, unsigned tile1,
                                            unsigned stride0) {
  unsigned long long ga = (unsigned long long)(size_t)gaddr;
  u32x4 g0 = { 1u,                                  // count=1, user descriptor
               lds_addr,                            // LDS byte address
               (unsigned)ga,                        // global_addr[31:0]
               (unsigned)((ga >> 32) & 0x1FFFFFFu) | (2u << 30) }; // [56:32]+type=2
  i32x8 g1 = { (int)(1u << 16),                     // data_size=1 (2 bytes)
               (int)((tdim0 & 0xFFFFu) << 16),      // tensor_dim0[15:0]
               (int)(((tdim0 >> 16) & 0xFFFFu) | ((tdim1 & 0xFFFFu) << 16)),
               (int)(((tdim1 >> 16) & 0xFFFFu) | ((tile0 & 0xFFFFu) << 16)),
               (int)(tile1 & 0xFFFFu),              // tile_dim1 (tile_dim2=0)
               (int)stride0,                        // tensor_dim0_stride[31:0]
               0, 0 };
  i32x4 z4 = { 0, 0, 0, 0 };
#if __clang_major__ >= 23
  i32x8 z8 = { 0, 0, 0, 0, 0, 0, 0, 0 };
  __builtin_amdgcn_tensor_load_to_lds(g0, g1, z4, z4, z8, 0);
#else
  __builtin_amdgcn_tensor_load_to_lds(g0, g1, z4, z4, 0);
#endif
}
#else
#define HAS_TDM 0
#endif

// ---- WMMA fragment loaders (contiguous k = 16h + e order, wave32) ------------------

// 16-bit A matrix 16x32 (MxK). lanes 0-15: M=lane, K=0..15;
// lanes 16-31: M=lane-16, K=16..31.
__device__ __forceinline__ v16h frag_a_load(const _Float16* p, int stride) {
  int lane = threadIdx.x & 31;
  int m  = lane & 15;
  int kb = (lane >> 4) << 4;
  v16h a;
#pragma unroll
  for (int e = 0; e < 16; ++e) {
    int k = kb + e;
    a[e] = p[m * stride + k];
  }
  return a;
}

// 16-bit B matrix 32x16 (KxN), source N-major: B(k,n) = p[n*stride + k].
__device__ __forceinline__ v16h frag_b_nmajor(const _Float16* p, int stride) {
  int lane = threadIdx.x & 31;
  int n  = lane & 15;
  int kb = (lane >> 4) << 4;
  v16h bf;
#pragma unroll
  for (int e = 0; e < 16; ++e) bf[e] = p[n * stride + kb + e];
  return bf;
}

// B layout, source K-major: B(k,n) = p[k*stride + n].
__device__ __forceinline__ v16h frag_b_kmajor(const _Float16* p, int stride) {
  int lane = threadIdx.x & 31;
  int n  = lane & 15;
  int kb = (lane >> 4) << 4;
  v16h bf;
#pragma unroll
  for (int e = 0; e < 16; ++e) bf[e] = p[(kb + e) * stride + n];
  return bf;
}

// ---------------------------------------------------------------------------
// Kernel 0: fp32 -> f16 conversion (one-time; weights are reused ~64x).
// ---------------------------------------------------------------------------
__global__ __launch_bounds__(256) void cvt_f32_f16(
    const float* __restrict__ in, _Float16* __restrict__ out, int n) {
  int i = (blockIdx.x * 256 + threadIdx.x) * 4;
  if (i < n) {
    float4 v = *(const float4*)(in + i);
    out[i + 0] = (_Float16)v.x;
    out[i + 1] = (_Float16)v.y;
    out[i + 2] = (_Float16)v.z;
    out[i + 3] = (_Float16)v.w;
  }
}

// ---------------------------------------------------------------------------
// Kernel 1/4: GEMM  C[M,N] = A[M,K] * W[N,K]^T, f16 in, f32 acc.
// 256 threads (8 waves), tile 128(M) x 128(N), K-step 32.
// Double-buffered async global->LDS staging: tile k+1 streams in while tile k
// feeds the WMMAs (s_wait_asynccnt 4 retires tile k; in-order per ISA 7.1).
// ---------------------------------------------------------------------------
template <bool HALF_OUT>
__global__ __launch_bounds__(256) void gemm_hh_wmma(
    const _Float16* __restrict__ A, const _Float16* __restrict__ W,
    void* __restrict__ Cout, int M, int N, int K) {
  const int LDT = 40;                        // padded stride in halfs (80 B)
  __shared__ __align__(16) _Float16 sA[2][128 * LDT];
  __shared__ __align__(16) _Float16 sW[2][128 * LDT];

  int tid  = threadIdx.x;
  int wave = tid >> 5;
  int m0 = blockIdx.x * 128;
  int n0 = blockIdx.y * 128;
  int NKT = K / 32;

  v8f acc[8] = {};

#if HAS_ASYNC_LDS
  auto stage_async = [&](int kt, int bufi) {  // 4 async b128 copies per thread
    int k0 = kt * 32;
    for (int i = tid; i < 512; i += 256) {
      int r = i >> 2, c = i & 3;
      async_copy16((const char*)A + ((size_t)(m0 + r) * K + k0) * 2 + c * 16,
                   (char*)sA[bufi] + r * 80 + c * 16);
      async_copy16((const char*)W + ((size_t)(n0 + r) * K + k0) * 2 + c * 16,
                   (char*)sW[bufi] + r * 80 + c * 16);
    }
  };
  stage_async(0, 0);                          // prologue: tile 0 in flight
#endif

  for (int kt = 0; kt < NKT; ++kt) {
    int buf = kt & 1;
    __syncthreads();                          // buf[(kt+1)&1] free to overwrite
#if HAS_ASYNC_LDS
    if (kt + 1 < NKT) {
      stage_async(kt + 1, buf ^ 1);
      __builtin_amdgcn_s_wait_asynccnt(4);    // tile kt done, kt+1 in flight
    } else {
      __builtin_amdgcn_s_wait_asynccnt(0);
    }
#else
    {
      int k0 = kt * 32;
      for (int i = tid; i < 512; i += 256) {
        int r = i >> 2, c = i & 3;
        *(uint4*)((char*)sA[buf] + r * 80 + c * 16) =
            *(const uint4*)((const char*)A + ((size_t)(m0 + r) * K + k0) * 2 + c * 16);
        *(uint4*)((char*)sW[buf] + r * 80 + c * 16) =
            *(const uint4*)((const char*)W + ((size_t)(n0 + r) * K + k0) * 2 + c * 16);
      }
    }
#endif
    __syncthreads();

    v16h af = frag_a_load(sA[buf] + wave * 16 * LDT, LDT);
#pragma unroll
    for (int g = 0; g < 8; ++g) {
      v16h bf = frag_b_nmajor(sW[buf] + g * 16 * LDT, LDT);
      acc[g] = WMMA_F32_F16(af, bf, acc[g]);
    }
  }

  int lane = tid & 31;
  int n  = lane & 15;
  int mh = (lane >> 4) << 3;
#pragma unroll
  for (int g = 0; g < 8; ++g)
#pragma unroll
    for (int r = 0; r < 8; ++r) {
      size_t idx = (size_t)(m0 + wave * 16 + mh + r) * N + n0 + g * 16 + n;
      if (HALF_OUT) ((_Float16*)Cout)[idx] = (_Float16)acc[g][r];
      else          ((float*)Cout)[idx]    = acc[g][r];
    }
}

// ---------------------------------------------------------------------------
// Kernel 2: per-token RoPE + flat-head RMSNorm + repack to (B,heads,S,128).
// grid = B*S, block = 256.
// ---------------------------------------------------------------------------
__device__ __forceinline__ float block_reduce_sum(float v, float* red) {
  int tid = threadIdx.x;
  red[tid] = v;
  __syncthreads();
  for (int s = 128; s > 0; s >>= 1) {
    if (tid < s) red[tid] += red[tid + s];
    __syncthreads();
  }
  float r = red[0];
  __syncthreads();
  return r;
}

__global__ __launch_bounds__(256) void rope_norm_repack(
    const _Float16* __restrict__ q16, const _Float16* __restrict__ k16,
    const _Float16* __restrict__ v16, const float* __restrict__ fc,
    const float* __restrict__ qw, const float* __restrict__ kw,
    _Float16* __restrict__ qh, _Float16* __restrict__ kh,
    _Float16* __restrict__ vh) {
  const int S = 2048;
  int token = blockIdx.x;
  int b = token >> 11, s = token & 2047;
  int tid = threadIdx.x;

  __shared__ float red[256];
  __shared__ float rq[2048];
  __shared__ float rk[512];

  float ssq = 0.f;
  for (int p = tid; p < 1024; p += 256) {
    int e0 = p * 2;
    int j  = (e0 & 127) >> 1;
    float x0 = (float)q16[(size_t)token * 2048 + e0];
    float x1 = (float)q16[(size_t)token * 2048 + e0 + 1];
    const float* f = fc + ((size_t)s * 64 + j) * 4;   // [[c,-s],[s,c]]
    float y0 = x0 * f[0] + x1 * f[1];
    float y1 = x0 * f[2] + x1 * f[3];
    rq[e0] = y0; rq[e0 + 1] = y1;
    ssq += y0 * y0 + y1 * y1;
  }
  ssq = block_reduce_sum(ssq, red);
  float qscale = rsqrtf(ssq * (1.0f / 2048.0f) + 1e-6f);

  float ssk;
  {
    int e0 = tid * 2;                                  // 256 pairs
    int j  = (e0 & 127) >> 1;
    float x0 = (float)k16[(size_t)token * 512 + e0];
    float x1 = (float)k16[(size_t)token * 512 + e0 + 1];
    const float* f = fc + ((size_t)s * 64 + j) * 4;
    float y0 = x0 * f[0] + x1 * f[1];
    float y1 = x0 * f[2] + x1 * f[3];
    rk[e0] = y0; rk[e0 + 1] = y1;
    ssk = y0 * y0 + y1 * y1;
  }
  ssk = block_reduce_sum(ssk, red);
  float kscale = rsqrtf(ssk * (1.0f / 512.0f) + 1e-6f);

  for (int e = tid; e < 2048; e += 256) {
    int h = e >> 7, d = e & 127;
    qh[(((size_t)b * 16 + h) * S + s) * 128 + d] = (_Float16)(rq[e] * qscale * qw[e]);
  }
  for (int e = tid; e < 512; e += 256) {
    int h = e >> 7, d = e & 127;
    kh[(((size_t)b * 4 + h) * S + s) * 128 + d] = (_Float16)(rk[e] * kscale * kw[e]);
    vh[(((size_t)b * 4 + h) * S + s) * 128 + d] = v16[(size_t)token * 512 + e];
  }
}

// ---------------------------------------------------------------------------
// Kernel 3: causal flash attention (GQA, n_rep=4).
// grid = B*16*(S/64), block = 128 (4 waves); each wave owns 16 query rows.
// K/V 32x128 tiles double-buffered through the Tensor Data Mover: wave 0
// issues tile kt+1's descriptors, then s_wait_tensorcnt 2 retires tile kt
// (TDM ops from one wave complete in order, ISA 7.1).
// ---------------------------------------------------------------------------
__global__ __launch_bounds__(128) void flash_attn_wmma(
    const _Float16* __restrict__ qh, const _Float16* __restrict__ kh,
    const _Float16* __restrict__ vh, _Float16* __restrict__ ao) {
  const int S = 2048;
  int bid  = blockIdx.x;
  int qblk = bid & 31;
  int h    = (bid >> 5) & 15;
  int b    = bid >> 9;
  int kvh  = h >> 2;

  int tid = threadIdx.x, wave = tid >> 5, lane = tid & 31;
  int q0 = qblk * 64 + wave * 16;
  int mrow = (lane >> 4) << 3;
  int ncol = lane & 15;

  __shared__ __align__(16) _Float16 sK[2][32 * 128];
  __shared__ __align__(16) _Float16 sV[2][32 * 128];
  __shared__ __align__(16) _Float16 sP[4][16 * 32];

  const _Float16* qptr = qh + (((size_t)b * 16 + h) * S + q0) * 128;
  v16h qa[4];
#pragma unroll
  for (int c = 0; c < 4; ++c) qa[c] = frag_a_load(qptr + c * 32, 128);

  v8f o[8] = {};
  float rmax[8], rsum[8];
#pragma unroll
  for (int r = 0; r < 8; ++r) { rmax[r] = -3.0e38f; rsum[r] = 0.f; }

  const _Float16* kbase = kh + ((size_t)b * 4 + kvh) * S * 128;
  const _Float16* vbase = vh + ((size_t)b * 4 + kvh) * S * 128;

  const float scale = 0.08838834764831845f;   // 1/sqrt(128)
  int ktiles = qblk * 2 + 2;

#if HAS_TDM
  unsigned ldsK0 = (unsigned)(size_t)(lds_f16_t*)sK[0];
  unsigned ldsK1 = (unsigned)(size_t)(lds_f16_t*)sK[1];
  unsigned ldsV0 = (unsigned)(size_t)(lds_f16_t*)sV[0];
  unsigned ldsV1 = (unsigned)(size_t)(lds_f16_t*)sV[1];
  if (wave == 0) {                            // prologue: tile 0 in flight
    tdm_load_2d(ldsK0, kbase, 128, S, 128, 32, 128);
    tdm_load_2d(ldsV0, vbase, 128, S, 128, 32, 128);
  }
#endif

  for (int kt = 0; kt < ktiles; ++kt) {
    int buf = kt & 1;
    __syncthreads();                          // buf^1 free to overwrite
#if HAS_TDM
    if (wave == 0) {                          // TDM ignores EXEC: wave-predicate
      if (kt + 1 < ktiles) {
        tdm_load_2d(buf ? ldsK0 : ldsK1,
                    kbase + (size_t)(kt + 1) * 32 * 128, 128, S, 128, 32, 128);
        tdm_load_2d(buf ? ldsV0 : ldsV1,
                    vbase + (size_t)(kt + 1) * 32 * 128, 128, S, 128, 32, 128);
        __builtin_amdgcn_s_wait_tensorcnt(2); // tile kt done, kt+1 in flight
      } else {
        __builtin_amdgcn_s_wait_tensorcnt(0);
      }
    }
#else
    {
      const uint4* kg = (const uint4*)(kbase + (size_t)kt * 32 * 128);
      const uint4* vg = (const uint4*)(vbase + (size_t)kt * 32 * 128);
      uint4* sk4 = (uint4*)sK[buf];
      uint4* sv4 = (uint4*)sV[buf];
      for (int i = tid; i < 512; i += 128) { sk4[i] = kg[i]; sv4[i] = vg[i]; }
    }
#endif
    __syncthreads();

    // Scores: Q(16x128) x K^T(128x32) -> two 16x16 f32 accumulators.
    v8f s0 = {}, s1 = {};
#pragma unroll
    for (int c = 0; c < 4; ++c) {
      v16h b0 = frag_b_nmajor(sK[buf] + c * 32, 128);
      s0 = WMMA_F32_F16(qa[c], b0, s0);
      v16h b1 = frag_b_nmajor(sK[buf] + 16 * 128 + c * 32, 128);
      s1 = WMMA_F32_F16(qa[c], b1, s1);
    }

    // Online softmax update (rows live across 16-lane half-waves).
#pragma unroll
    for (int r = 0; r < 8; ++r) {
      int q  = q0 + mrow + r;
      int kk = kt * 32 + ncol;
      float e0 = (kk      <= q) ? s0[r] * scale : -3.0e38f;
      float e1 = (kk + 16 <= q) ? s1[r] * scale : -3.0e38f;
      float t = fmaxf(e0, e1);
#pragma unroll
      for (int m = 1; m < 16; m <<= 1) t = fmaxf(t, __shfl_xor(t, m, 16));
      float nm = fmaxf(rmax[r], t);
      float al = __expf(rmax[r] - nm);
      float p0 = (kk      <= q) ? __expf(s0[r] * scale - nm) : 0.f;
      float p1 = (kk + 16 <= q) ? __expf(s1[r] * scale - nm) : 0.f;
      float rs = p0 + p1;
#pragma unroll
      for (int m = 1; m < 16; m <<= 1) rs += __shfl_xor(rs, m, 16);
      rsum[r] = rsum[r] * al + rs;
      rmax[r] = nm;
#pragma unroll
      for (int g = 0; g < 8; ++g) o[g][r] *= al;
      sP[wave][(mrow + r) * 32 + ncol]      = (_Float16)p0;
      sP[wave][(mrow + r) * 32 + 16 + ncol] = (_Float16)p1;
    }

    // PV: P(16x32) x V(32x128).
    v16h pa = frag_a_load(&sP[wave][0], 32);
#pragma unroll
    for (int g = 0; g < 8; ++g) {
      v16h vf = frag_b_kmajor(sV[buf] + g * 16, 128);
      o[g] = WMMA_F32_F16(pa, vf, o[g]);
    }
  }

#pragma unroll
  for (int r = 0; r < 8; ++r) {
    float inv = 1.0f / rsum[r];
    int q = q0 + mrow + r;
#pragma unroll
    for (int g = 0; g < 8; ++g)
      ao[((size_t)b * S + q) * 2048 + h * 128 + g * 16 + ncol] =
          (_Float16)(o[g][r] * inv);
  }
}

// ---------------------------------------------------------------------------
// Host-side launch
// ---------------------------------------------------------------------------
extern "C" void kernel_launch(void* const* d_in, const int* in_sizes, int n_in,
                              void* d_out, int out_size, void* d_ws, size_t ws_size,
                              hipStream_t stream) {
  const int B = 2, S = 2048, DIM = 2048, H = 16, KV = 4, HD = 128;
  const int M = B * S;                        // 4096
  const int NQ = H * HD;                      // 2048
  const int NKV = KV * HD;                    // 512

  const float* x  = (const float*)d_in[0];
  const float* fc = (const float*)d_in[1];
  const float* wq = (const float*)d_in[2];
  const float* wk = (const float*)d_in[3];
  const float* wv = (const float*)d_in[4];
  const float* wo = (const float*)d_in[5];
  const float* qw = (const float*)d_in[6];
  const float* kw = (const float*)d_in[7];
  float* out = (float*)d_out;

  // Workspace layout (~100 MB).
  char* ws = (char*)d_ws;
  size_t off = 0;
  _Float16* xh  = (_Float16*)(ws + off); off += (size_t)M  * DIM * 2;
  _Float16* wqh = (_Float16*)(ws + off); off += (size_t)NQ * DIM * 2;
  _Float16* wkh = (_Float16*)(ws + off); off += (size_t)NKV * DIM * 2;
  _Float16* wvh = (_Float16*)(ws + off); off += (size_t)NKV * DIM * 2;
  _Float16* woh = (_Float16*)(ws + off); off += (size_t)DIM * NQ * 2;
  _Float16* q16 = (_Float16*)(ws + off); off += (size_t)M * NQ  * 2;
  _Float16* k16 = (_Float16*)(ws + off); off += (size_t)M * NKV * 2;
  _Float16* v16 = (_Float16*)(ws + off); off += (size_t)M * NKV * 2;
  _Float16* qhp = (_Float16*)(ws + off); off += (size_t)M * NQ  * 2;
  _Float16* khp = (_Float16*)(ws + off); off += (size_t)M * NKV * 2;
  _Float16* vhp = (_Float16*)(ws + off); off += (size_t)M * NKV * 2;
  _Float16* aoh = (_Float16*)(ws + off); off += (size_t)M * NQ  * 2;
  (void)ws_size; (void)in_sizes; (void)n_in; (void)out_size;

  dim3 blk128(128), blk256(256);

  // 0) One-time f16 conversions (activations + weights).
  cvt_f32_f16<<<dim3((M * DIM) / 1024), blk256, 0, stream>>>(x,  xh,  M * DIM);
  cvt_f32_f16<<<dim3((NQ * DIM) / 1024), blk256, 0, stream>>>(wq, wqh, NQ * DIM);
  cvt_f32_f16<<<dim3((NKV * DIM) / 1024), blk256, 0, stream>>>(wk, wkh, NKV * DIM);
  cvt_f32_f16<<<dim3((NKV * DIM) / 1024), blk256, 0, stream>>>(wv, wvh, NKV * DIM);
  cvt_f32_f16<<<dim3((DIM * NQ) / 1024), blk256, 0, stream>>>(wo, woh, DIM * NQ);

  // 1) Q/K/V projections (f16 out).
  gemm_hh_wmma<true><<<dim3(M / 128, NQ  / 128), blk256, 0, stream>>>(xh, wqh, q16, M, NQ,  DIM);
  gemm_hh_wmma<true><<<dim3(M / 128, NKV / 128), blk256, 0, stream>>>(xh, wkh, k16, M, NKV, DIM);
  gemm_hh_wmma<true><<<dim3(M / 128, NKV / 128), blk256, 0, stream>>>(xh, wvh, v16, M, NKV, DIM);

  // 2) RoPE + flat-head RMSNorm + head-major repack.
  rope_norm_repack<<<dim3(M), blk256, 0, stream>>>(q16, k16, v16, fc, qw, kw, qhp, khp, vhp);

  // 3) Causal flash attention (GQA), double-buffered TDM K/V tiles.
  flash_attn_wmma<<<dim3(B * H * (S / 64)), blk128, 0, stream>>>(qhp, khp, vhp, aoh);

  // 4) Output projection (f32 out to d_out).
  gemm_hh_wmma<false><<<dim3(M / 128, DIM / 128), blk256, 0, stream>>>(aoh, woh, out, M, DIM, NQ);
}
